// NONLocalBlock2D_82188494176758
// MI455X (gfx1250) — matrix operand order repaired and checked by reading, not compiled, not run on hardware
//
#include <hip/hip_runtime.h>
#include <hip/hip_bf16.h>

// ---------------- problem constants ----------------
#define B_   8
#define C_   256
#define CI_  128
#define H_   64
#define W_   64
#define N_   (H_*W_)          // 4096
#define M_   ((H_/2)*(W_/2))  // 1024

typedef __attribute__((ext_vector_type(16))) __bf16 bf16x16;
typedef __attribute__((ext_vector_type(8)))  float  f32x8;
typedef __attribute__((ext_vector_type(4)))  unsigned uint32x4;
typedef __attribute__((ext_vector_type(8)))  int      int32x8;
typedef __attribute__((ext_vector_type(4)))  int      int32x4;

union FragU { bf16x16 v; unsigned u[8]; };

#if defined(__has_builtin)
#if __has_builtin(__builtin_amdgcn_tensor_load_to_lds)
#define HAVE_TDM 1
#endif
#endif

// ---- bf16 helpers (raw ushort storage) ----
static __device__ __forceinline__ unsigned short f2bf(float f) {
    unsigned u = __builtin_bit_cast(unsigned, f);
    u += 0x7FFFu + ((u >> 16) & 1u);          // round-to-nearest-even
    return (unsigned short)(u >> 16);
}
static __device__ __forceinline__ float bf2f(unsigned short h) {
    unsigned u = ((unsigned)h) << 16;
    return __builtin_bit_cast(float, u);
}

// ---- WMMA fragment loaders (wave32 layouts per CDNA5 ISA 7.12.2) ----
// A fragment: 16(M) x 32(K); element(m,k) = src[m*ld + k]; ld even.
static __device__ __forceinline__ bf16x16 load_frag_a(const unsigned short* src, int ld) {
    int lane = threadIdx.x & 31;
    int hx   = lane >> 4;
    int m    = lane & 15;
    const unsigned* p = (const unsigned*)(src + (size_t)m * ld);
    FragU f;
#pragma unroll
    for (int v = 0; v < 4; ++v) f.u[v]     = p[hx*4 + v];        // K = hx*8 + 2v
#pragma unroll
    for (int v = 0; v < 4; ++v) f.u[4 + v] = p[8 + hx*4 + v];    // K = 16 + hx*8 + 2v
    return f.v;
}
// B fragment: 32(K) x 16(N); element(n,k) = src[n*ld + k]; ld even.
static __device__ __forceinline__ bf16x16 load_frag_b(const unsigned short* src, int ld) {
    int lane = threadIdx.x & 31;
    int hx   = lane >> 4;
    int n    = lane & 15;
    const unsigned* p = (const unsigned*)(src + (size_t)n * ld);
    FragU f;
#pragma unroll
    for (int j = 0; j < 8; ++j) f.u[j] = p[(j >> 2)*8 + hx*4 + (j & 3)];  // K = (j>>2)*16 + hx*8 + 2(j&3)
    return f.v;
}
static __device__ __forceinline__ f32x8 wmma_bf16(bf16x16 a, bf16x16 b, f32x8 c) {
    return __builtin_amdgcn_wmma_f32_16x16x32_bf16(false, a, false, b, (short)0, c, false, false);
}
static __device__ __forceinline__ f32x8 zero8() {
    f32x8 z;
#pragma unroll
    for (int i = 0; i < 8; ++i) z[i] = 0.0f;
    return z;
}
static __device__ __forceinline__ float rowmax16(float v) {
#pragma unroll
    for (int m = 8; m >= 1; m >>= 1) v = fmaxf(v, __shfl_xor(v, m, 32));
    return v;
}
static __device__ __forceinline__ float rowsum16(float v) {
#pragma unroll
    for (int m = 8; m >= 1; m >>= 1) v += __shfl_xor(v, m, 32);
    return v;
}

#ifdef HAVE_TDM
// Build a 2-D tile D# per CDNA5 ISA sec 8.3-8.6 and issue TENSOR_LOAD_TO_LDS.
// data_size = 2 bytes (code 1); padding inserted on LDS writes for bank spread.
// clang-23 toolchain: 6-arg form (g0 v4u, g1 v8i, g2 v4i, g3 v4i, g4 v8i, cpol).
static __device__ __forceinline__ void tdm_load_2d(
    unsigned lds_addr, const void* gptr,
    unsigned tensor_d0, unsigned tensor_d1,
    unsigned tile_d0, unsigned tile_d1,
    unsigned stride0, unsigned pad_interval, unsigned pad_amount) {
    unsigned long long ga = (unsigned long long)(uintptr_t)gptr;
    uint32x4 g0; int32x8 g1; int32x4 g2, g3; int32x8 g4;
    g0[0] = 1u;                                     // count=1, user mode
    g0[1] = lds_addr;                               // lds_addr [63:32]
    g0[2] = (unsigned)ga;                           // global_addr low
    g0[3] = (unsigned)((ga >> 32) & 0x01FFFFFFu) | (2u << 30);  // addr hi | type=2
    g1[0] = (int)((1u << 16) | (1u << 20) | (pad_interval << 22) | (pad_amount << 25));
    g1[1] = (int)((tensor_d0 & 0xFFFFu) << 16);                  // tensor_dim0 lo16
    g1[2] = (int)(((tensor_d0 >> 16) & 0xFFFFu) | ((tensor_d1 & 0xFFFFu) << 16));
    g1[3] = (int)(((tensor_d1 >> 16) & 0xFFFFu) | ((tile_d0 & 0xFFFFu) << 16));
    g1[4] = (int)(tile_d1 & 0xFFFFu);                            // tile_dim1, tile_dim2=0
    g1[5] = (int)stride0;                                        // dim0 stride lo32
    g1[6] = 0; g1[7] = 0;                                        // stride hi, dim1 stride
    g2[0] = g2[1] = g2[2] = g2[3] = 0;
    g3[0] = g3[1] = g3[2] = g3[3] = 0;
#pragma unroll
    for (int i = 0; i < 8; ++i) g4[i] = 0;
    __builtin_amdgcn_tensor_load_to_lds(g0, g1, g2, g3, g4, 0);
}
#endif

// ---------------- kernel 0: weights -> bf16 ----------------
__global__ void wconv_kernel(const float* wt, const float* wp, const float* wg,
                             const float* wz, unsigned short* wbf) {
    int idx = blockIdx.x * blockDim.x + threadIdx.x;
    const int per = CI_ * C_;
    if (idx >= 4 * per) return;
    const float* src = (idx < per) ? wt : (idx < 2*per) ? wp : (idx < 3*per) ? wg : wz;
    wbf[idx] = f2bf(src[idx % per]);
}

// ---------------- kernel 1: fused theta/phi/g 1x1 conv (WMMA) ----------------
__global__ void __launch_bounds__(128)
conv_tpg_kernel(const float* __restrict__ x, const unsigned short* __restrict__ wbf,
                const float* __restrict__ b_theta, const float* __restrict__ b_phi,
                const float* __restrict__ b_g,
                unsigned short* thetaT, unsigned short* phiF, unsigned short* gF) {
    __shared__ __align__(16) unsigned short xt[64][40];
    int tid  = threadIdx.x;
    int wave = tid >> 5;
    int lane = tid & 31;
    int hx   = lane >> 4;
    int lc   = lane & 15;
    int n0   = blockIdx.x * 64;
    int b    = blockIdx.y;
    int conv = blockIdx.z;

    const unsigned short* w = wbf + (size_t)conv * CI_ * C_;
    const float* bias = (conv == 0) ? b_theta : (conv == 1) ? b_phi : b_g;
    unsigned short* outp = (conv == 0) ? thetaT : (conv == 1) ? phiF : gF;

    f32x8 acc[8];
#pragma unroll
    for (int t = 0; t < 8; ++t) acc[t] = zero8();

    for (int kc = 0; kc < C_; kc += 32) {
#pragma unroll
        for (int p = 0; p < 4; ++p) {
            int c  = p * 8 + (tid >> 4);
            int nn = (tid & 15) * 4;
            const float* src = x + ((size_t)b * C_ + kc + c) * (size_t)N_ + n0 + nn;
#pragma unroll
            for (int q = 0; q < 4; ++q) xt[nn + q][c] = f2bf(src[q]);
        }
        __syncthreads();
        bf16x16 aF = load_frag_a(&xt[wave * 16][0], 40);
        bf16x16 bFr[8];
#pragma unroll
        for (int t = 0; t < 8; ++t) bFr[t] = load_frag_b(w + (size_t)(t * 16) * C_ + kc, C_);
#pragma unroll
        for (int t = 0; t < 8; ++t) acc[t] = wmma_bf16(aF, bFr[t], acc[t]);
        __syncthreads();
    }
#pragma unroll
    for (int t = 0; t < 8; ++t) {
        int col = t * 16 + lc;
        float bv = bias[col];
#pragma unroll
        for (int r = 0; r < 8; ++r) {
            int n = n0 + wave * 16 + r + 8 * hx;
            outp[((size_t)b * N_ + n) * CI_ + col] = f2bf(acc[t][r] + bv);
        }
    }
}

// ---------------- kernel 2: 2x2 max-pool phi & g ----------------
__global__ void pool_kernel(const unsigned short* __restrict__ phiF,
                            const unsigned short* __restrict__ gF,
                            unsigned short* phi_p, unsigned short* g_pT) {
    int idx = blockIdx.x * blockDim.x + threadIdx.x;
    if (idx >= B_ * M_ * CI_) return;
    {
        int i = idx % CI_; int m = (idx / CI_) % M_; int b = idx / (CI_ * M_);
        int hp = m >> 5, wp = m & 31;
        int n00 = (hp * 2) * W_ + wp * 2;
        const unsigned short* p = phiF + (size_t)b * N_ * CI_ + i;
        float v = fmaxf(fmaxf(bf2f(p[(size_t)n00 * CI_]),        bf2f(p[(size_t)(n00 + 1) * CI_])),
                        fmaxf(bf2f(p[(size_t)(n00 + W_) * CI_]), bf2f(p[(size_t)(n00 + W_ + 1) * CI_])));
        phi_p[idx] = f2bf(v);
    }
    {
        int m = idx % M_; int i = (idx / M_) % CI_; int b = idx / (M_ * CI_);
        int hp = m >> 5, wp = m & 31;
        int n00 = (hp * 2) * W_ + wp * 2;
        const unsigned short* q = gF + (size_t)b * N_ * CI_ + i;
        float v = fmaxf(fmaxf(bf2f(q[(size_t)n00 * CI_]),        bf2f(q[(size_t)(n00 + 1) * CI_])),
                        fmaxf(bf2f(q[(size_t)(n00 + W_) * CI_]), bf2f(q[(size_t)(n00 + W_ + 1) * CI_])));
        g_pT[idx] = f2bf(v);
    }
}

// ---------------- kernel 3: fused attention, TDM double-buffered tiles ----------------
__global__ void __launch_bounds__(128)
attn_kernel(const unsigned short* __restrict__ thetaT,
            const unsigned short* __restrict__ phi_p,
            const unsigned short* __restrict__ g_pT,
            unsigned short* yT) {
    // phi tile: 32 m-rows x 128 i (pad rows to 136 -> TDM pad 4 dwords / 64 dwords)
    // g tile  : 128 i-rows x 32 m (pad rows to 40  -> TDM pad 4 dwords / 16 dwords)
    __shared__ __align__(16) unsigned short phi_s[2][32][136];
    __shared__ __align__(16) unsigned short g_s[2][128][40];
    __shared__ __align__(16) unsigned short pbuf[4][16][40];
    int tid  = threadIdx.x;
    int wave = tid >> 5;
    int lane = tid & 31;
    int hx   = lane >> 4;
    int lc   = lane & 15;
    int b    = blockIdx.y;
    int n0   = blockIdx.x * 64 + wave * 16;

    const unsigned short* thb  = thetaT + ((size_t)b * N_ + n0) * CI_;
    const unsigned short* phib = phi_p  + (size_t)b * M_ * CI_;
    const unsigned short* gb   = g_pT   + (size_t)b * CI_ * M_;

    bf16x16 thA[4];
#pragma unroll
    for (int k = 0; k < 4; ++k) thA[k] = load_frag_a(thb + k * 32, CI_);

    f32x8 yacc[8];
#pragma unroll
    for (int t = 0; t < 8; ++t) yacc[t] = zero8();
    float rmax[8], rsum[8];
#pragma unroll
    for (int r = 0; r < 8; ++r) { rmax[r] = -__builtin_inff(); rsum[r] = 0.0f; }

    const int NSTEP = M_ / 32;    // 32 tiles of 32 m-columns

#ifdef HAVE_TDM
    if (wave == 0) {   // prefetch tile 0 (EXEC-independent DMA; one wave owns TENSORcnt)
        tdm_load_2d((unsigned)(uintptr_t)&phi_s[0][0][0], phib,
                    CI_, M_, CI_, 32, CI_, /*pad_int=*/5, /*pad_amt=*/3);
        tdm_load_2d((unsigned)(uintptr_t)&g_s[0][0][0], gb,
                    M_, CI_, 32, CI_, M_, /*pad_int=*/3, /*pad_amt=*/3);
    }
#endif

    for (int step = 0; step < NSTEP; ++step) {
        int cur = step & 1;
        int mt  = step * 32;
#ifdef HAVE_TDM
        if (wave == 0) {
            if (step + 1 < NSTEP) {
                tdm_load_2d((unsigned)(uintptr_t)&phi_s[cur ^ 1][0][0],
                            phib + (size_t)(mt + 32) * CI_,
                            CI_, M_, CI_, 32, CI_, 5, 3);
                tdm_load_2d((unsigned)(uintptr_t)&g_s[cur ^ 1][0][0],
                            gb + (size_t)(mt + 32),
                            M_, CI_, 32, CI_, M_, 3, 3);
                __builtin_amdgcn_s_wait_tensorcnt(2);   // current tile's 2 loads done
            } else {
                __builtin_amdgcn_s_wait_tensorcnt(0);
            }
        }
        __syncthreads();
#else
        // cooperative staged copy (fallback, no TDM builtin)
        for (int ch = tid; ch < 512; ch += 128) {       // phi: 32 rows x 16 chunks(16B)
            int m = ch >> 4, c = ch & 15;
            *(uint4*)&phi_s[cur][m][c * 8] =
                *(const uint4*)(phib + (size_t)(mt + m) * CI_ + c * 8);
        }
        for (int ch = tid; ch < 512; ch += 128) {       // g: 128 rows x 4 chunks(16B)
            int i = ch >> 2, c = ch & 3;
            *(uint4*)&g_s[cur][i][c * 8] =
                *(const uint4*)(gb + (size_t)i * M_ + mt + c * 8);
        }
        __syncthreads();
#endif
        // ---- scores S = theta x phi^T for 32 m-columns ----
        f32x8 S0 = zero8(), S1 = zero8();
#pragma unroll
        for (int k = 0; k < 4; ++k) {
            S0 = wmma_bf16(thA[k], load_frag_b(&phi_s[cur][0][k * 32], 136), S0);
            S1 = wmma_bf16(thA[k], load_frag_b(&phi_s[cur][16][k * 32], 136), S1);
        }
        // ---- online softmax (per-row stats live in the 16 lanes of each half) ----
#pragma unroll
        for (int r = 0; r < 8; ++r) {
            float tmax = rowmax16(fmaxf(S0[r], S1[r]));
            float newm = fmaxf(rmax[r], tmax);
            float corr = __expf(rmax[r] - newm);
            float p0 = __expf(S0[r] - newm);
            float p1 = __expf(S1[r] - newm);
            rsum[r] = rsum[r] * corr + rowsum16(p0 + p1);
            rmax[r] = newm;
#pragma unroll
            for (int t = 0; t < 8; ++t) yacc[t][r] *= corr;
            int row = r + 8 * hx;
            pbuf[wave][row][lc]      = f2bf(p0);
            pbuf[wave][row][16 + lc] = f2bf(p1);
        }
        asm volatile("s_wait_dscnt 0" ::: "memory");    // per-wave LDS RAW fence
        bf16x16 pA = load_frag_a(&pbuf[wave][0][0], 40);
        // ---- y += P x g^T ----
#pragma unroll
        for (int it = 0; it < 8; ++it) {
            yacc[it] = wmma_bf16(pA, load_frag_b(&g_s[cur][it * 16][0], 40), yacc[it]);
        }
        __syncthreads();
    }
#pragma unroll
    for (int r = 0; r < 8; ++r) {
        float inv = 1.0f / rsum[r];
#pragma unroll
        for (int t = 0; t < 8; ++t) yacc[t][r] *= inv;
    }
    unsigned short* yb = yT + ((size_t)b * N_ + n0) * CI_;
#pragma unroll
    for (int t = 0; t < 8; ++t) {
        int col = t * 16 + lc;
#pragma unroll
        for (int r = 0; r < 8; ++r) {
            int row = r + 8 * hx;
            yb[(size_t)row * CI_ + col] = f2bf(yacc[t][r]);
        }
    }
}

// ---------------- kernel 4: z-conv (WMMA) -> wy f32 ----------------
__global__ void __launch_bounds__(128)
convz_kernel(const unsigned short* __restrict__ wzbf, const unsigned short* __restrict__ yT,
             const float* __restrict__ b_z, float* wy) {
    int wave = threadIdx.x >> 5;
    int lane = threadIdx.x & 31;
    int hx   = lane >> 4;
    int lc   = lane & 15;
    int b  = blockIdx.z;
    int o0 = blockIdx.y * 64 + wave * 16;
    int n0 = blockIdx.x * 16;

    f32x8 acc = zero8();
#pragma unroll
    for (int k = 0; k < 4; ++k) {
        bf16x16 aF = load_frag_a(wzbf + (size_t)o0 * CI_ + k * 32, CI_);
        bf16x16 bF = load_frag_b(yT + ((size_t)b * N_ + n0) * CI_ + k * 32, CI_);
        acc = wmma_bf16(aF, bF, acc);
    }
    int n = n0 + lc;
#pragma unroll
    for (int r = 0; r < 8; ++r) {
        int o = o0 + r + 8 * hx;
        wy[((size_t)b * C_ + o) * N_ + n] = acc[r] + b_z[o];
    }
}

// ---------------- kernel 5: per-channel batch stats ----------------
__global__ void __launch_bounds__(256)
bnstats_kernel(const float* __restrict__ wy, float* meanv, float* istdv) {
    int o = blockIdx.x;
    int tid = threadIdx.x;
    float s = 0.0f, q = 0.0f;
    for (int b = 0; b < B_; ++b) {
        const float* p = wy + ((size_t)b * C_ + o) * N_;
        for (int n = tid; n < N_; n += 256) { float v = p[n]; s += v; q += v * v; }
    }
    __shared__ float ss[256], qq[256];
    ss[tid] = s; qq[tid] = q;
    __syncthreads();
    for (int st = 128; st > 0; st >>= 1) {
        if (tid < st) { ss[tid] += ss[tid + st]; qq[tid] += qq[tid + st]; }
        __syncthreads();
    }
    if (tid == 0) {
        float inv = 1.0f / (float)(B_ * N_);
        float m = ss[0] * inv;
        float var = qq[0] * inv - m * m;
        meanv[o] = m;
        istdv[o] = rsqrtf(var + 1e-5f);
    }
}

// ---------------- kernel 6: BN + residual ----------------
__global__ void bnapply_kernel(const float* __restrict__ wy, const float* __restrict__ x,
                               const float* __restrict__ gamma, const float* __restrict__ beta,
                               const float* __restrict__ meanv, const float* __restrict__ istdv,
                               float* out) {
    size_t idx = (size_t)blockIdx.x * blockDim.x + threadIdx.x;
    if (idx >= (size_t)B_ * C_ * N_) return;
    int o = (int)((idx / N_) % C_);
    out[idx] = gamma[o] * (wy[idx] - meanv[o]) * istdv[o] + beta[o] + x[idx];
}

// ---------------- launcher ----------------
extern "C" void kernel_launch(void* const* d_in, const int* in_sizes, int n_in,
                              void* d_out, int out_size, void* d_ws, size_t ws_size,
                              hipStream_t stream) {
    const float* x       = (const float*)d_in[0];
    const float* w_theta = (const float*)d_in[1];
    const float* b_theta = (const float*)d_in[2];
    const float* w_phi   = (const float*)d_in[3];
    const float* b_phi   = (const float*)d_in[4];
    const float* w_g     = (const float*)d_in[5];
    const float* b_g     = (const float*)d_in[6];
    const float* w_z     = (const float*)d_in[7];
    const float* b_z     = (const float*)d_in[8];
    const float* gamma   = (const float*)d_in[9];
    const float* beta    = (const float*)d_in[10];
    float* out = (float*)d_out;

    char* base = (char*)d_ws;
    size_t off = 0;
    auto alloc = [&](size_t bytes) -> void* {
        void* r = base + off;
        off = (off + bytes + 255) & ~(size_t)255;
        return r;
    };
    unsigned short* wbf    = (unsigned short*)alloc((size_t)4 * CI_ * C_ * 2);
    unsigned short* thetaT = (unsigned short*)alloc((size_t)B_ * N_ * CI_ * 2);
    unsigned short* phiF   = (unsigned short*)alloc((size_t)B_ * N_ * CI_ * 2);
    unsigned short* gF     = (unsigned short*)alloc((size_t)B_ * N_ * CI_ * 2);
    unsigned short* phi_p  = (unsigned short*)alloc((size_t)B_ * M_ * CI_ * 2);
    unsigned short* g_pT   = (unsigned short*)alloc((size_t)B_ * M_ * CI_ * 2);
    unsigned short* yT     = (unsigned short*)alloc((size_t)B_ * N_ * CI_ * 2);
    float*          wy     = (float*)alloc((size_t)B_ * C_ * N_ * 4);
    float*          meanv  = (float*)alloc(C_ * 4);
    float*          istdv  = (float*)alloc(C_ * 4);
    unsigned short* wzbf   = wbf + (size_t)3 * CI_ * C_;

    wconv_kernel<<<(4 * CI_ * C_ + 255) / 256, 256, 0, stream>>>(w_theta, w_phi, w_g, w_z, wbf);
    conv_tpg_kernel<<<dim3(N_ / 64, B_, 3), 128, 0, stream>>>(
        x, wbf, b_theta, b_phi, b_g, thetaT, phiF, gF);
    pool_kernel<<<((B_ * M_ * CI_) + 255) / 256, 256, 0, stream>>>(phiF, gF, phi_p, g_pT);
    attn_kernel<<<dim3(N_ / 64, B_), 128, 0, stream>>>(thetaT, phi_p, g_pT, yT);
    convz_kernel<<<dim3(N_ / 16, C_ / 64, B_), 128, 0, stream>>>(wzbf, yT, b_z, wy);
    bnstats_kernel<<<C_, 256, 0, stream>>>(wy, meanv, istdv);
    bnapply_kernel<<<(int)(((size_t)B_ * C_ * N_ + 255) / 256), 256, 0, stream>>>(
        wy, x, gamma, beta, meanv, istdv, out);
}
